// OptimizedConformationalConsistencyLoss_15607911153865
// MI455X (gfx1250) — matrix operand order repaired and checked by reading, not compiled, not run on hardware
//
#include <hip/hip_runtime.h>
#include <hip/hip_bf16.h>

typedef __attribute__((ext_vector_type(16))) _Float16 v16h;
typedef __attribute__((ext_vector_type(8)))  _Float16 v8h;
typedef __attribute__((ext_vector_type(8)))  float    v8f;

constexpr int HD  = 256;   // hidden channels
constexpr int NF  = 2048;  // fragments
constexpr int TMR = 128;   // rows per block in the GEMM kernel (4 waves x 32)

// ---------------------------------------------------------------------------
// WMMA fragment loaders (f16, 16x16x32).
// A operand, 16-bit 16x32 (ISA 7.12.2): lane l (l<16): row M=l, K in
// {k0..k0+7, k0+16..k0+23}; lane l>=16: same row, K in {k0+8..15, k0+24..31}.
// B operand, 16-bit 32x16: lane l holds column N=l&15; same K split as A:
// lanes 0-15 K in {k0..k0+7, k0+16..k0+23}, lanes 16-31 {k0+8..15, k0+24..31}.
// ---------------------------------------------------------------------------
__device__ __forceinline__ v16h afrag(const _Float16* row, int k0, int half) {
  const _Float16* p = row + k0 + half * 8;
  v8h lo = *(const v8h*)p;
  v8h hi = *(const v8h*)(p + 16);
  v16h r;
#pragma unroll
  for (int i = 0; i < 8; ++i) { r[i] = lo[i]; r[i + 8] = hi[i]; }
  return r;
}
__device__ __forceinline__ v16h bfrag(const _Float16* col, int k0, int half) {
  const _Float16* p = col + k0 + half * 8;
  v8h lo = *(const v8h*)p;
  v8h hi = *(const v8h*)(p + 16);
  v16h r;
#pragma unroll
  for (int i = 0; i < 8; ++i) { r[i] = lo[i]; r[i + 8] = hi[i]; }
  return r;
}

// full-wave (32 lane) sum
__device__ __forceinline__ float wsum(float x) {
#pragma unroll
  for (int m = 16; m >= 1; m >>= 1) x += __shfl_xor(x, m, 32);
  return x;
}
// sum within each 16-lane half (xor masks < 16 never cross halves)
__device__ __forceinline__ float hsum(float x) {
#pragma unroll
  for (int m = 1; m < 16; m <<= 1) x += __shfl_xor(x, m, 32);
  return x;
}

// ---------------------------------------------------------------------------
// K0: zero accumulator region
// ---------------------------------------------------------------------------
__global__ void k_zero(float* p, int n) {
  int i = blockIdx.x * blockDim.x + threadIdx.x;
  for (; i < n; i += gridDim.x * blockDim.x) p[i] = 0.f;
}

// ---------------------------------------------------------------------------
// K0b: W1,W2 -> f16 transposed (column-contiguous for B operand);
//      ssq = || colsum(Wv) ||^2   (vproj[i,:] = vmag[i] * colsum(Wv))
// ---------------------------------------------------------------------------
__global__ void k_prepw(const float* __restrict__ W1, const float* __restrict__ W2,
                        const float* __restrict__ Wv,
                        _Float16* __restrict__ W1T, _Float16* __restrict__ W2T,
                        float* __restrict__ ssq) {
  int id = blockIdx.x * 256 + threadIdx.x;        // grid = 256 blocks -> HD*HD
  int k = id >> 8, n = id & 255;
  W1T[n * HD + k] = (_Float16)W1[k * HD + n];
  W2T[n * HD + k] = (_Float16)W2[k * HD + n];
  if (id < HD) {
    float s = 0.f;
    for (int kk = 0; kk < HD; ++kk) s += Wv[kk * HD + id];
    atomicAdd(ssq, s * s);
  }
}

// ---------------------------------------------------------------------------
// K1: fused  sc-combine -> GEMM1 -> +b1 -> LayerNorm -> SiLU -> GEMM2 -> +b2
//     -> proj store + segment-sum atomics.
// One wave owns 32 rows (two A fragments) so each B fragment feeds 2 WMMAs,
// halving operand-delivery bandwidth per FLOP.  Block = 4 waves = 128 rows.
// ---------------------------------------------------------------------------
__global__ __launch_bounds__(128, 1)
void k_gemm_proj(const float* __restrict__ ss, const float* __restrict__ sl,
                 const _Float16* __restrict__ W1T, const float* __restrict__ b1,
                 const float* __restrict__ lng, const float* __restrict__ lnb,
                 const _Float16* __restrict__ W2T, const float* __restrict__ b2,
                 const int* __restrict__ seg,
                 float* __restrict__ proj, float* __restrict__ grp,
                 float* __restrict__ cnt, int N) {
  __shared__ _Float16 sh[TMR * HD];                 // 64 KB, 16 KB slice / wave
  const int wave = threadIdx.x >> 5;                // 0..3
  const int lane = threadIdx.x & 31;
  const int r    = lane & 15;
  const int half = lane >> 4;
  const int row0 = blockIdx.x * TMR + wave * 32;
  _Float16* mysh = sh + wave * 32 * HD;

  // ---- stage sc = 0.5*(ss+sl) as f16 into this wave's LDS slice ----
  {
    const int c0 = lane * 8;
#pragma unroll 4
    for (int rr = 0; rr < 32; ++rr) {
      int g = row0 + rr;
      v8h out;
      if (g < N) {
        const float4 a0 = *(const float4*)(ss + (size_t)g * HD + c0);
        const float4 a1 = *(const float4*)(ss + (size_t)g * HD + c0 + 4);
        const float4 c0v = *(const float4*)(sl + (size_t)g * HD + c0);
        const float4 c1v = *(const float4*)(sl + (size_t)g * HD + c0 + 4);
        out[0] = (_Float16)(0.5f * (a0.x + c0v.x));
        out[1] = (_Float16)(0.5f * (a0.y + c0v.y));
        out[2] = (_Float16)(0.5f * (a0.z + c0v.z));
        out[3] = (_Float16)(0.5f * (a0.w + c0v.w));
        out[4] = (_Float16)(0.5f * (a1.x + c1v.x));
        out[5] = (_Float16)(0.5f * (a1.y + c1v.y));
        out[6] = (_Float16)(0.5f * (a1.z + c1v.z));
        out[7] = (_Float16)(0.5f * (a1.w + c1v.w));
      } else {
#pragma unroll
        for (int i = 0; i < 8; ++i) out[i] = (_Float16)0.f;
      }
      *(v8h*)(mysh + rr * HD + c0) = out;
    }
  }

  const v8f vzero = {0.f, 0.f, 0.f, 0.f, 0.f, 0.f, 0.f, 0.f};
  v8f accA[16], accB[16];
#pragma unroll
  for (int nt = 0; nt < 16; ++nt) { accA[nt] = vzero; accB[nt] = vzero; }

  // ---- GEMM1: h = sc @ W1  (one B frag -> two WMMAs) ----
  for (int kb = 0; kb < HD / 32; ++kb) {
    const int k0 = kb * 32;
    v16h a0 = afrag(mysh + r * HD, k0, half);
    v16h a1 = afrag(mysh + (16 + r) * HD, k0, half);
#pragma unroll
    for (int nt = 0; nt < 16; ++nt) {
      v16h b = bfrag(W1T + (nt * 16 + r) * HD, k0, half);
      accA[nt] = __builtin_amdgcn_wmma_f32_16x16x32_f16(
          false, a0, false, b, (short)0, accA[nt], false, false);
      accB[nt] = __builtin_amdgcn_wmma_f32_16x16x32_f16(
          false, a1, false, b, (short)0, accB[nt], false, false);
    }
  }

  // ---- +b1 ----
#pragma unroll
  for (int nt = 0; nt < 16; ++nt) {
    float bb = b1[nt * 16 + r];
#pragma unroll
    for (int v = 0; v < 8; ++v) { accA[nt][v] += bb; accB[nt][v] += bb; }
  }

  // ---- LayerNorm stats: row m = v + half*8 lives in vgpr v of one half ----
  float muA[8], rsA[8], muB[8], rsB[8];
#pragma unroll
  for (int v = 0; v < 8; ++v) {
    float sA = 0.f, qA = 0.f, sB = 0.f, qB = 0.f;
#pragma unroll
    for (int nt = 0; nt < 16; ++nt) {
      float xA = accA[nt][v]; sA += xA; qA += xA * xA;
      float xB = accB[nt][v]; sB += xB; qB += xB * xB;
    }
    sA = hsum(sA); qA = hsum(qA); sB = hsum(sB); qB = hsum(qB);
    float mA = sA * (1.0f / HD), mB = sB * (1.0f / HD);
    muA[v] = mA; rsA[v] = rsqrtf(qA * (1.0f / HD) - mA * mA + 1e-5f);
    muB[v] = mB; rsB[v] = rsqrtf(qB * (1.0f / HD) - mB * mB + 1e-5f);
  }

  // ---- normalize + SiLU, write h back to LDS as f16 (same-wave only) ----
#pragma unroll
  for (int nt = 0; nt < 16; ++nt) {
    int n = nt * 16 + r;
    float g_ = lng[n], be_ = lnb[n];
#pragma unroll
    for (int v = 0; v < 8; ++v) {
      float xA = (accA[nt][v] - muA[v]) * rsA[v] * g_ + be_;
      float yA = xA / (1.0f + __expf(-xA));
      mysh[(v + half * 8) * HD + n] = (_Float16)yA;
      float xB = (accB[nt][v] - muB[v]) * rsB[v] * g_ + be_;
      float yB = xB / (1.0f + __expf(-xB));
      mysh[(16 + v + half * 8) * HD + n] = (_Float16)yB;
    }
    accA[nt] = vzero; accB[nt] = vzero;
  }

  // ---- GEMM2: proj = h @ W2 ----
  for (int kb = 0; kb < HD / 32; ++kb) {
    const int k0 = kb * 32;
    v16h a0 = afrag(mysh + r * HD, k0, half);
    v16h a1 = afrag(mysh + (16 + r) * HD, k0, half);
#pragma unroll
    for (int nt = 0; nt < 16; ++nt) {
      v16h b = bfrag(W2T + (nt * 16 + r) * HD, k0, half);
      accA[nt] = __builtin_amdgcn_wmma_f32_16x16x32_f16(
          false, a0, false, b, (short)0, accA[nt], false, false);
      accB[nt] = __builtin_amdgcn_wmma_f32_16x16x32_f16(
          false, a1, false, b, (short)0, accB[nt], false, false);
    }
  }

  // ---- +b2, store proj, segment sums ----
  int  segA[8], segB[8];
  bool okA[8],  okB[8];
#pragma unroll
  for (int v = 0; v < 8; ++v) {
    int gA = row0 + v + half * 8;
    int gB = gA + 16;
    okA[v] = (gA < N);  segA[v] = okA[v] ? seg[gA] : 0;
    okB[v] = (gB < N);  segB[v] = okB[v] ? seg[gB] : 0;
    if (okA[v] && r == 0) atomicAdd(&cnt[segA[v]], 1.0f);
    if (okB[v] && r == 0) atomicAdd(&cnt[segB[v]], 1.0f);
  }
#pragma unroll
  for (int nt = 0; nt < 16; ++nt) {
    int n = nt * 16 + r;
    float bb = b2[n];
#pragma unroll
    for (int v = 0; v < 8; ++v) {
      if (okA[v]) {
        float val = accA[nt][v] + bb;
        int g = row0 + v + half * 8;
        proj[(size_t)g * HD + n] = val;
        atomicAdd(&grp[(size_t)segA[v] * HD + n], val);
      }
      if (okB[v]) {
        float val = accB[nt][v] + bb;
        int g = row0 + 16 + v + half * 8;
        proj[(size_t)g * HD + n] = val;
        atomicAdd(&grp[(size_t)segB[v] * HD + n], val);
      }
    }
  }
}

// ---------------------------------------------------------------------------
// K2: group mean + row-normalized f16 copy (for similarity WMMA)
// ---------------------------------------------------------------------------
__global__ __launch_bounds__(256, 1)
void k_groupmean(const float* __restrict__ grp, const float* __restrict__ cnt,
                 float* __restrict__ mean, _Float16* __restrict__ nrmh) {
  __shared__ float red[8];
  int f = blockIdx.x, c = threadIdx.x;
  float inv = 1.0f / fmaxf(cnt[f], 1.0f);
  float v = grp[(size_t)f * HD + c] * inv;
  mean[(size_t)f * HD + c] = v;
  float q = wsum(v * v);
  if ((threadIdx.x & 31) == 0) red[threadIdx.x >> 5] = q;
  __syncthreads();
  float n2 = 0.f;
#pragma unroll
  for (int i = 0; i < 8; ++i) n2 += red[i];
  float nrm = fmaxf(sqrtf(n2), 1e-12f);
  nrmh[(size_t)f * HD + c] = (_Float16)(v / nrm);
}

// ---------------------------------------------------------------------------
// K3: adaptive target MLP, one wave per fragment (hidden = 32 = lane count)
// ---------------------------------------------------------------------------
__global__ __launch_bounds__(256, 1)
void k_target(const float* __restrict__ mean, const float* __restrict__ Wt1,
              const float* __restrict__ bt1, const float* __restrict__ Wt2,
              const float* __restrict__ bt2, float* __restrict__ tsum) {
  int wave = threadIdx.x >> 5, lane = threadIdx.x & 31;
  int f = blockIdx.x * 8 + wave;
  const float* mrow = mean + (size_t)f * HD;
  float h = bt1[lane];
  for (int k = 0; k < HD; ++k) h = fmaf(mrow[k], Wt1[k * 32 + lane], h);
  float s = h / (1.0f + __expf(-h));
  float t = wsum(s * Wt2[lane]);
  if (lane == 0) {
    t += bt2[0];
    float sg = 1.0f / (1.0f + __expf(-t));
    sg = fminf(fmaxf(sg, 0.2f), 0.8f);
    atomicAdd(tsum, sg);
  }
}

// ---------------------------------------------------------------------------
// K4: similarity matrix (nrm @ nrm^T) via WMMA + fused huber, upper triangle.
//     Grid = 136 upper-triangular 128x128 block tiles.
// ---------------------------------------------------------------------------
__global__ __launch_bounds__(256, 1)
void k_sim(const _Float16* __restrict__ nrmh, const float* __restrict__ scal,
           float* __restrict__ hub) {
  int idx = blockIdx.x, bi = 0;
  while (idx >= (16 - bi)) { idx -= (16 - bi); ++bi; }
  int bj = bi + idx;

  const int wave = threadIdx.x >> 5, lane = threadIdx.x & 31;
  const int r = lane & 15, half = lane >> 4;
  const float target = scal[0] * (1.0f / NF);

  const v8f vzero = {0.f, 0.f, 0.f, 0.f, 0.f, 0.f, 0.f, 0.f};
  v8f acc[8];
#pragma unroll
  for (int nt = 0; nt < 8; ++nt) acc[nt] = vzero;

  const _Float16* Arow = nrmh + (size_t)(bi * 128 + wave * 16 + r) * HD;
  for (int kb = 0; kb < HD / 32; ++kb) {
    int k0 = kb * 32;
    v16h a = afrag(Arow, k0, half);
#pragma unroll
    for (int nt = 0; nt < 8; ++nt) {
      v16h b = bfrag(nrmh + (size_t)(bj * 128 + nt * 16 + r) * HD, k0, half);
      acc[nt] = __builtin_amdgcn_wmma_f32_16x16x32_f16(
          false, a, false, b, (short)0, acc[nt], false, false);
    }
  }

  float ls = 0.f;
  int mgb = bi * 128 + wave * 16 + half * 8;
#pragma unroll
  for (int nt = 0; nt < 8; ++nt) {
    int ng = bj * 128 + nt * 16 + r;
#pragma unroll
    for (int v = 0; v < 8; ++v) {
      if (ng > mgb + v) {                        // strict upper triangle
        float d = fabsf(acc[nt][v] - target);
        ls += (d < 0.1f) ? 0.5f * d * d : 0.1f * (d - 0.05f);
      }
    }
  }
  ls = wsum(ls);
  if (lane == 0) atomicAdd(hub, ls);
}

// ---------------------------------------------------------------------------
// K5: per-atom squared distance to fragment mean, one wave per atom
// ---------------------------------------------------------------------------
__global__ __launch_bounds__(256, 1)
void k_var(const float* __restrict__ proj, const float* __restrict__ mean,
           const int* __restrict__ seg, float* __restrict__ vars, int N) {
  int wave = threadIdx.x >> 5, lane = threadIdx.x & 31;
  int i = blockIdx.x * 8 + wave;
  if (i >= N) return;
  int s = seg[i];
  const float* p = proj + (size_t)i * HD + lane * 8;
  const float* m = mean + (size_t)s * HD + lane * 8;
  float4 p0 = *(const float4*)p, p1 = *(const float4*)(p + 4);
  float4 m0 = *(const float4*)m, m1 = *(const float4*)(m + 4);
  float d = 0.f, t;
  t = p0.x - m0.x; d += t * t;  t = p0.y - m0.y; d += t * t;
  t = p0.z - m0.z; d += t * t;  t = p0.w - m0.w; d += t * t;
  t = p1.x - m1.x; d += t * t;  t = p1.y - m1.y; d += t * t;
  t = p1.z - m1.z; d += t * t;  t = p1.w - m1.w; d += t * t;
  d = wsum(d);
  if (lane == 0) atomicAdd(&vars[s], d);
}

// ---------------------------------------------------------------------------
// K6: vmag streaming pass (the 614 MB read), one wave per atom
// ---------------------------------------------------------------------------
__global__ __launch_bounds__(256, 1)
void k_vmag(const float* __restrict__ vs, const float* __restrict__ vl,
            const int* __restrict__ seg, float* __restrict__ vsum,
            float* __restrict__ vsq, int N) {
  int wave = threadIdx.x >> 5, lane = threadIdx.x & 31;
  int i = blockIdx.x * 8 + wave;
  if (i >= N) return;
  float nm = 0.f;
#pragma unroll
  for (int a = 0; a < 3; ++a) {
    size_t base = (size_t)i * (3 * HD) + a * HD + lane * 8;
    float4 s0 = *(const float4*)(vs + base), s1 = *(const float4*)(vs + base + 4);
    float4 l0 = *(const float4*)(vl + base), l1 = *(const float4*)(vl + base + 4);
    float q = 0.f, t;
    t = 0.5f * (s0.x + l0.x); q += t * t;  t = 0.5f * (s0.y + l0.y); q += t * t;
    t = 0.5f * (s0.z + l0.z); q += t * t;  t = 0.5f * (s0.w + l0.w); q += t * t;
    t = 0.5f * (s1.x + l1.x); q += t * t;  t = 0.5f * (s1.y + l1.y); q += t * t;
    t = 0.5f * (s1.z + l1.z); q += t * t;  t = 0.5f * (s1.w + l1.w); q += t * t;
    q = wsum(q);
    nm += sqrtf(q);
  }
  float vm = nm * (1.0f / 3.0f);
  if (lane == 0) {
    int sg = seg[i];
    atomicAdd(&vsum[sg], vm);
    atomicAdd(&vsq[sg], vm * vm);
  }
}

// ---------------------------------------------------------------------------
// K7: per-fragment finalize (intra std for scalar + vector branches)
// ---------------------------------------------------------------------------
__global__ void k_fragfin(const float* __restrict__ cnt, const float* __restrict__ vars,
                          const float* __restrict__ vsum, const float* __restrict__ vsq,
                          const float* __restrict__ ssq,
                          float* intra, float* valid, float* vec) {
  int f = blockIdx.x * 256 + threadIdx.x;
  if (f >= NF) return;
  float c = cnt[f];
  if (c >= 2.0f) {
    float v  = vars[f] / c;
    float mv = vsum[f] / c;
    float vm = fmaxf(vsq[f] / c - mv * mv, 0.f);
    atomicAdd(intra, sqrtf(v + 1e-8f));
    atomicAdd(vec,   sqrtf(ssq[0] * vm + 1e-8f));
    atomicAdd(valid, 1.0f);
  }
}

// ---------------------------------------------------------------------------
// K8: scalar combine
// ---------------------------------------------------------------------------
__global__ void k_final(const float* __restrict__ scal, float* __restrict__ out) {
  if (threadIdx.x == 0 && blockIdx.x == 0) {
    float valid = scal[3];
    float intra = valid > 0.f ? scal[2] / valid : 0.f;
    float vloss = valid > 0.f ? scal[4] / valid : 0.f;
    float gsl   = scal[1] * (1.0f / 2096128.0f);   // F*(F-1)/2
    out[0] = 0.05f * (0.3f * intra + 0.7f * gsl + 0.1f * vloss);
  }
}

// ---------------------------------------------------------------------------
extern "C" void kernel_launch(void* const* d_in, const int* in_sizes, int n_in,
                              void* d_out, int out_size, void* d_ws, size_t ws_size,
                              hipStream_t stream) {
  const float* ss  = (const float*)d_in[0];
  const float* sl  = (const float*)d_in[1];
  const float* vs  = (const float*)d_in[2];
  const float* vl  = (const float*)d_in[3];
  const float* W1  = (const float*)d_in[4];
  const float* b1  = (const float*)d_in[5];
  const float* lng = (const float*)d_in[6];
  const float* lnb = (const float*)d_in[7];
  const float* W2  = (const float*)d_in[8];
  const float* b2  = (const float*)d_in[9];
  const float* Wv  = (const float*)d_in[10];
  const float* Wt1 = (const float*)d_in[11];
  const float* bt1 = (const float*)d_in[12];
  const float* Wt2 = (const float*)d_in[13];
  const float* bt2 = (const float*)d_in[14];
  const int*   seg = (const int*)d_in[15];
  const int N = in_sizes[15];

  // workspace layout (all offsets 256B aligned)
  char* w = (char*)d_ws;
  size_t off = 0;
  float* scal = (float*)(w + off); off += 256;                 // [0]tsum [1]hub [2]intra [3]valid [4]vec [5]ssq
  float* grp  = (float*)(w + off); off += (size_t)NF * HD * 4;
  float* mean = (float*)(w + off); off += (size_t)NF * HD * 4;
  float* cnt  = (float*)(w + off); off += NF * 4;
  float* vars = (float*)(w + off); off += NF * 4;
  float* vsum = (float*)(w + off); off += NF * 4;
  float* vsq  = (float*)(w + off); off += NF * 4;
  _Float16* W1T  = (_Float16*)(w + off); off += HD * HD * 2;
  _Float16* W2T  = (_Float16*)(w + off); off += HD * HD * 2;
  _Float16* nrmh = (_Float16*)(w + off); off += (size_t)NF * HD * 2;
  float* proj = (float*)(w + off); off += (size_t)N * HD * 4;

  int zcount = 64 + NF * HD * 2 + NF * 4;   // scal..vsq contiguous region
  k_zero<<<1024, 256, 0, stream>>>(scal, zcount);
  k_prepw<<<256, 256, 0, stream>>>(W1, W2, Wv, W1T, W2T, &scal[5]);

  int nb = (N + TMR - 1) / TMR;
  k_gemm_proj<<<nb, 128, 0, stream>>>(ss, sl, W1T, b1, lng, lnb, W2T, b2, seg,
                                      proj, grp, cnt, N);
  k_groupmean<<<NF, 256, 0, stream>>>(grp, cnt, mean, nrmh);
  k_target<<<NF / 8, 256, 0, stream>>>(mean, Wt1, bt1, Wt2, bt2, &scal[0]);
  k_sim<<<136, 256, 0, stream>>>(nrmh, &scal[0], &scal[1]);

  int ab = (N + 7) / 8;
  k_var<<<ab, 256, 0, stream>>>(proj, mean, seg, vars, N);
  k_vmag<<<ab, 256, 0, stream>>>(vs, vl, seg, vsum, vsq, N);
  k_fragfin<<<NF / 256, 256, 0, stream>>>(cnt, vars, vsum, vsq, &scal[5],
                                          &scal[2], &scal[3], &scal[4]);
  k_final<<<1, 64, 0, stream>>>(scal, (float*)d_out);
}
